// SepConv3DOverBatch_44890998177899
// MI455X (gfx1250) — matrix operand order repaired and checked by reading, not compiled, not run on hardware
//
#include <hip/hip_runtime.h>

// Problem constants (from the reference)
#define T_ 8
#define H_ 96
#define W_ 160
#define C_ 256
#define F_ 256
#define TILE_M 32           // output w-positions per block
#define AS_STRIDE 41        // fp32-fallback LDS stride (floats); coprime with 64 banks
#define AROWS 264           // bf16 A-tile row stride in shorts (528 B, 16B-aligned, m*528%256=16m)

typedef __attribute__((ext_vector_type(2)))  float  v2f;
typedef __attribute__((ext_vector_type(8)))  float  v8f;
typedef __attribute__((ext_vector_type(16))) __bf16 v16bf;

struct U32B { uint4 lo, hi; };   // 32 bytes

__device__ __forceinline__ unsigned short f32_to_bf16_rne(float f) {
    unsigned u = __builtin_bit_cast(unsigned, f);
    u += 0x7FFFu + ((u >> 16) & 1u);            // round-to-nearest-even
    return (unsigned short)(u >> 16);
}
__device__ __forceinline__ float bf16_bits_to_f32(unsigned short h) {
    return __builtin_bit_cast(float, (unsigned)h << 16);
}
__device__ __forceinline__ v16bf make_v16bf(uint4 a, uint4 b) {
    U32B t{a, b};
    return __builtin_bit_cast(v16bf, t);
}

// ---------------------------------------------------------------------------
// Pre-pass: split Wp (fp32 [C][F]) into hi/lo bf16, packed in WMMA B-fragment
// order: idx = ((kchunk*F + n)*32 + k%32).  One (k,n) per thread.
// ---------------------------------------------------------------------------
__global__ void wp_to_bf16x2_kernel(const float* __restrict__ Wp,
                                    unsigned short* __restrict__ bh,
                                    unsigned short* __restrict__ bl) {
    const int k = blockIdx.x;        // 0..255 (C)
    const int n = threadIdx.x;       // 0..255 (F)
    const float v = Wp[(size_t)k * F_ + n];
    const unsigned short h = f32_to_bf16_rne(v);
    const unsigned short l = f32_to_bf16_rne(v - bf16_bits_to_f32(h));
    const size_t idx = ((size_t)((k >> 5) * F_ + n)) * 32 + (k & 31);
    bh[idx] = h;
    bl[idx] = l;
}

// ---------------------------------------------------------------------------
// Main fused kernel (bf16x3 error-compensated GEMM on v_wmma_f32_16x16x32_bf16)
// ---------------------------------------------------------------------------
__launch_bounds__(256)
__global__ void SepConv3D_bf16x3_kernel(const float* __restrict__ x,
                                        const float* __restrict__ Wd,
                                        const unsigned short* __restrict__ Bh,
                                        const unsigned short* __restrict__ Bl,
                                        float* __restrict__ out)
{
    // A tile split hi/lo bf16, [m][k] with padded row stride (shorts).
    __shared__ __align__(16) unsigned short AhS[TILE_M * AROWS];  // 16,896 B
    __shared__ __align__(16) unsigned short AlS[TILE_M * AROWS];  // 16,896 B

    const int tid = threadIdx.x;
    const int w0  = blockIdx.x * TILE_M;
    const int h   = blockIdx.y;
    const int t   = blockIdx.z;

    // ---------------- Phase 1: depthwise 3x3x3 (thread = channel) ----------------
    {
        const int c = tid;
        float acc[TILE_M];
        #pragma unroll
        for (int m = 0; m < TILE_M; ++m) acc[m] = 0.0f;

        for (int kt = 0; kt < 3; ++kt) {
            const int tt = t + kt - 1;
            if (tt < 0 || tt >= T_) continue;
            for (int kh = 0; kh < 3; ++kh) {
                const int hh = h + kh - 1;
                if (hh < 0 || hh >= H_) continue;
                const float* row = x + ((size_t)(tt * H_ + hh) * W_) * C_ + c;
                const float wk0 = Wd[(((size_t)kt * 3 + kh) * 3 + 0) * C_ + c];
                const float wk1 = Wd[(((size_t)kt * 3 + kh) * 3 + 1) * C_ + c];
                const float wk2 = Wd[(((size_t)kt * 3 + kh) * 3 + 2) * C_ + c];

                float r[TILE_M + 2];
                #pragma unroll
                for (int wi = 0; wi < TILE_M + 2; ++wi) {
                    const int ww = w0 + wi - 1;
                    r[wi] = (ww >= 0 && ww < W_) ? row[(size_t)ww * C_] : 0.0f;
                }
                #pragma unroll
                for (int m = 0; m < TILE_M; ++m) {
                    acc[m] = fmaf(r[m + 0], wk0, acc[m]);
                    acc[m] = fmaf(r[m + 1], wk1, acc[m]);
                    acc[m] = fmaf(r[m + 2], wk2, acc[m]);
                }
            }
        }
        // Split fp32 -> (hi, lo) bf16 and store A tile [m][k=c]
        #pragma unroll
        for (int m = 0; m < TILE_M; ++m) {
            const unsigned short hb = f32_to_bf16_rne(acc[m]);
            const unsigned short lb = f32_to_bf16_rne(acc[m] - bf16_bits_to_f32(hb));
            AhS[m * AROWS + c] = hb;
            AlS[m * AROWS + c] = lb;
        }
    }
    __syncthreads();

    // ------- Phase 2: GEMM [32 x 256] x [256 x 256] + ReLU, bf16x3 WMMA -------
    {
        const int wave = tid >> 5;
        const int lane = tid & 31;
        const int hi   = lane >> 4;      // lane half selects K sub-ranges per ISA layout
        const int lmod = lane & 15;      // M (A) / N (B, C, D)
        const int n0   = wave * 32;      // two 16-wide N tiles per wave

        const v8f vzero = {0.f, 0.f, 0.f, 0.f, 0.f, 0.f, 0.f, 0.f};
        v8f acc[2][2];
        acc[0][0] = vzero; acc[0][1] = vzero; acc[1][0] = vzero; acc[1][1] = vzero;

        #pragma unroll 2
        for (int kc = 0; kc < C_ / 32; ++kc) {       // 8 chunks of K=32
            v16bf Ahf[2], Alf[2], Bhf[2], Blf[2];

            // A fragments from LDS: lanes 0-15 take K=kc*32+{0..7,16..23},
            // lanes 16-31 take K=kc*32+{8..15,24..31}  (VGPR0-3 / VGPR4-7)
            #pragma unroll
            for (int mt = 0; mt < 2; ++mt) {
                const int off = (mt * 16 + lmod) * AROWS + kc * 32 + hi * 8;
                Ahf[mt] = make_v16bf(*(const uint4*)(AhS + off),
                                     *(const uint4*)(AhS + off + 16));
                Alf[mt] = make_v16bf(*(const uint4*)(AlS + off),
                                     *(const uint4*)(AlS + off + 16));
            }
            // B fragments from pre-packed global: lane reads K=kc*32+{8h..8h+7, 16+8h..16+8h+7}
            #pragma unroll
            for (int nt = 0; nt < 2; ++nt) {
                const size_t off = ((size_t)(kc * F_ + n0 + nt * 16 + lmod)) * 32 + hi * 8;
                Bhf[nt] = make_v16bf(*(const uint4*)(Bh + off),
                                     *(const uint4*)(Bh + off + 16));
                Blf[nt] = make_v16bf(*(const uint4*)(Bl + off),
                                     *(const uint4*)(Bl + off + 16));
            }

            // D += Ah*Bh + Ah*Bl + Al*Bh   (error-compensated fp32-accurate GEMM)
            #pragma unroll
            for (int mt = 0; mt < 2; ++mt) {
                #pragma unroll
                for (int nt = 0; nt < 2; ++nt) {
                    acc[mt][nt] = __builtin_amdgcn_wmma_f32_16x16x32_bf16(
                        false, Ahf[mt], false, Bhf[nt], (short)0, acc[mt][nt], false, false);
                    acc[mt][nt] = __builtin_amdgcn_wmma_f32_16x16x32_bf16(
                        false, Ahf[mt], false, Blf[nt], (short)0, acc[mt][nt], false, false);
                    acc[mt][nt] = __builtin_amdgcn_wmma_f32_16x16x32_bf16(
                        false, Alf[mt], false, Bhf[nt], (short)0, acc[mt][nt], false, false);
                }
            }
        }

        // Store with fused ReLU; C/D layout: VGPR r -> M = r + 8*(lane/16), N = lane%16
        #pragma unroll
        for (int mt = 0; mt < 2; ++mt) {
            #pragma unroll
            for (int nt = 0; nt < 2; ++nt) {
                #pragma unroll
                for (int r = 0; r < 8; ++r) {
                    const int m = mt * 16 + r + 8 * hi;
                    const int n = n0 + nt * 16 + lmod;
                    float v = acc[mt][nt][r];
                    v = v > 0.0f ? v : 0.0f;
                    out[((size_t)(t * H_ + h) * W_ + (w0 + m)) * F_ + n] = v;
                }
            }
        }
    }
}

// ---------------------------------------------------------------------------
// Fallback: native fp32 WMMA (16x16x4) fused kernel — used if d_ws too small.
// ---------------------------------------------------------------------------
__launch_bounds__(256)
__global__ void SepConv3D_f32_kernel(const float* __restrict__ x,
                                     const float* __restrict__ Wd,
                                     const float* __restrict__ Wp,
                                     float* __restrict__ out)
{
    __shared__ float As[C_ * AS_STRIDE];

    const int tid = threadIdx.x;
    const int w0  = blockIdx.x * TILE_M;
    const int h   = blockIdx.y;
    const int t   = blockIdx.z;

    {
        const int c = tid;
        float acc[TILE_M];
        #pragma unroll
        for (int m = 0; m < TILE_M; ++m) acc[m] = 0.0f;

        for (int kt = 0; kt < 3; ++kt) {
            const int tt = t + kt - 1;
            if (tt < 0 || tt >= T_) continue;
            for (int kh = 0; kh < 3; ++kh) {
                const int hh = h + kh - 1;
                if (hh < 0 || hh >= H_) continue;
                const float* row = x + ((size_t)(tt * H_ + hh) * W_) * C_ + c;
                const float wk0 = Wd[(((size_t)kt * 3 + kh) * 3 + 0) * C_ + c];
                const float wk1 = Wd[(((size_t)kt * 3 + kh) * 3 + 1) * C_ + c];
                const float wk2 = Wd[(((size_t)kt * 3 + kh) * 3 + 2) * C_ + c];
                float r[TILE_M + 2];
                #pragma unroll
                for (int wi = 0; wi < TILE_M + 2; ++wi) {
                    const int ww = w0 + wi - 1;
                    r[wi] = (ww >= 0 && ww < W_) ? row[(size_t)ww * C_] : 0.0f;
                }
                #pragma unroll
                for (int m = 0; m < TILE_M; ++m) {
                    acc[m] = fmaf(r[m + 0], wk0, acc[m]);
                    acc[m] = fmaf(r[m + 1], wk1, acc[m]);
                    acc[m] = fmaf(r[m + 2], wk2, acc[m]);
                }
            }
        }
        #pragma unroll
        for (int m = 0; m < TILE_M; ++m)
            As[c * AS_STRIDE + m] = acc[m];
    }
    __syncthreads();

    {
        const int wave = tid >> 5;
        const int lane = tid & 31;
        const int hi   = lane >> 4;
        const int lmod = lane & 15;
        const int n0   = wave * 32;

        const v8f vzero = {0.f, 0.f, 0.f, 0.f, 0.f, 0.f, 0.f, 0.f};
        v8f acc[2][2];
        acc[0][0] = vzero; acc[0][1] = vzero; acc[1][0] = vzero; acc[1][1] = vzero;

        #pragma unroll 4
        for (int k0 = 0; k0 < C_; k0 += 4) {
            const int ka = k0 + 2 * hi;
            v2f a0, a1;
            a0.x = As[(ka + 0) * AS_STRIDE + lmod];
            a0.y = As[(ka + 1) * AS_STRIDE + lmod];
            a1.x = As[(ka + 0) * AS_STRIDE + 16 + lmod];
            a1.y = As[(ka + 1) * AS_STRIDE + 16 + lmod];
            const float* bp = Wp + (size_t)ka * F_ + n0 + lmod;
            v2f b0, b1;
            b0.x = bp[0];
            b0.y = bp[F_];
            b1.x = bp[16];
            b1.y = bp[F_ + 16];
            acc[0][0] = __builtin_amdgcn_wmma_f32_16x16x4_f32(false, a0, false, b0,
                                                              (short)0, acc[0][0], false, false);
            acc[0][1] = __builtin_amdgcn_wmma_f32_16x16x4_f32(false, a0, false, b1,
                                                              (short)0, acc[0][1], false, false);
            acc[1][0] = __builtin_amdgcn_wmma_f32_16x16x4_f32(false, a1, false, b0,
                                                              (short)0, acc[1][0], false, false);
            acc[1][1] = __builtin_amdgcn_wmma_f32_16x16x4_f32(false, a1, false, b1,
                                                              (short)0, acc[1][1], false, false);
        }

        #pragma unroll
        for (int mt = 0; mt < 2; ++mt) {
            #pragma unroll
            for (int nt = 0; nt < 2; ++nt) {
                #pragma unroll
                for (int r = 0; r < 8; ++r) {
                    const int m = mt * 16 + r + 8 * hi;
                    const int n = n0 + nt * 16 + lmod;
                    float v = acc[mt][nt][r];
                    v = v > 0.0f ? v : 0.0f;
                    out[((size_t)(t * H_ + h) * W_ + (w0 + m)) * F_ + n] = v;
                }
            }
        }
    }
}

extern "C" void kernel_launch(void* const* d_in, const int* in_sizes, int n_in,
                              void* d_out, int out_size, void* d_ws, size_t ws_size,
                              hipStream_t stream) {
    const float* x  = (const float*)d_in[0];   // [8,96,160,256] fp32
    const float* Wd = (const float*)d_in[1];   // [3,3,3,1,256]  fp32
    const float* Wp = (const float*)d_in[2];   // [256,256]      fp32
    float* out      = (float*)d_out;           // [8,96,160,256] fp32

    dim3 grid(W_ / TILE_M, H_, T_);            // 5 x 96 x 8 = 3840 blocks

    const size_t need = (size_t)2 * C_ * F_ * sizeof(unsigned short);  // 256 KB
    if (ws_size >= need) {
        unsigned short* bh = (unsigned short*)d_ws;
        unsigned short* bl = bh + (size_t)C_ * F_;
        wp_to_bf16x2_kernel<<<dim3(C_), dim3(F_), 0, stream>>>(Wp, bh, bl);
        SepConv3D_bf16x3_kernel<<<grid, 256, 0, stream>>>(x, Wd, bh, bl, out);
    } else {
        SepConv3D_f32_kernel<<<grid, 256, 0, stream>>>(x, Wd, Wp, out);
    }
}
